// HmmLm_70265664962569
// MI455X (gfx1250) — matrix operand order repaired and checked by reading, not compiled, not run on hardware
//
#include <hip/hip_runtime.h>
#include <math.h>

// Problem constants (from reference)
#define CC 256     // HMM states
#define HH 256     // hidden
#define VV 10000   // vocab
#define NN 16      // batch
#define TT 65      // time
#define PP (NN*TT) // flattened (n,t) token positions = 1040 = 65*16
#define CHUNKS 25            // V split into 25 chunks for the online LSE
#define TILES_PER_CHUNK 25   // 25*25 = 625 = V/16 exactly

typedef __attribute__((ext_vector_type(16))) __bf16 v16bf;
typedef __attribute__((ext_vector_type(8)))  float  v8f;

union Frag32B { uint4 u[2]; v16bf v; };

// round-to-nearest-even f32 -> bf16 (bit trick; avoids relying on fp conversions)
__device__ __forceinline__ __bf16 f2bf(float x) {
  union { float f; unsigned u; } a; a.f = x;
  unsigned r = a.u + 0x7FFFu + ((a.u >> 16) & 1u);
  union { unsigned short s; __bf16 b; } o;
  o.s = (unsigned short)(r >> 16);
  return o.b;
}

__global__ void cvt_f32_to_bf16_kernel(const float* __restrict__ src,
                                       __bf16* __restrict__ dst, int n) {
  int i = blockIdx.x * blockDim.x + threadIdx.x;
  if (i < n) dst[i] = f2bf(src[i]);
}

// A fragment: 16x32 bf16 tile of row-major A[M][lda].
// lane&15 = M row; lanes<16 hold K = kb..kb+7 and kb+16..kb+23; lanes>=16 the +8 offsets.
__device__ __forceinline__ v16bf load_frag_a(const __bf16* __restrict__ A, int lda,
                                             int mbase, int kb, int lane) {
  int m = mbase + (lane & 15);
  int sel = (lane < 16) ? 0 : 8;
  const __bf16* p = A + (size_t)m * lda + kb;
  Frag32B f;
  f.u[0] = *reinterpret_cast<const uint4*>(p + sel);
  f.u[1] = *reinterpret_cast<const uint4*>(p + 16 + sel);
  return f.v;
}

// B fragment: 32x16 bf16 where B[k,n] = W[row(n), k] (W rows are K-contiguous).
// Each lane supplies output column n = (its row); lanes<16 hold K = kb..kb+7 and kb+16..kb+23; lanes>=16 the +8 offsets.
__device__ __forceinline__ v16bf load_frag_b_row(const __bf16* __restrict__ W, int ldw,
                                                 int row, int kb, int lane) {
  int koff = kb + ((lane < 16) ? 0 : 8);
  const __bf16* p = W + (size_t)row * ldw + koff;
  Frag32B f;
  f.u[0] = reinterpret_cast<const uint4*>(p)[0];
  f.u[1] = reinterpret_cast<const uint4*>(p)[2];
  return f.v;
}

// Y = [relu](A @ W^T + bias) [+ resid]; one wave per 16x16 tile, K-loop of WMMA.
__global__ void gemm256_bf16_wmma_kernel(const __bf16* __restrict__ A,
                                         const __bf16* __restrict__ W,
                                         const float* __restrict__ bias,
                                         const float* __restrict__ resid,
                                         float* __restrict__ Yf32,
                                         __bf16* __restrict__ Yb16,
                                         int Nn, int K, int do_relu) {
  int ntn = Nn >> 4;
  int mbase = (blockIdx.x / ntn) << 4;
  int nbase = (blockIdx.x % ntn) << 4;
  int lane = threadIdx.x;
  v8f c = {};
  for (int kb = 0; kb < K; kb += 32) {
    v16bf a = load_frag_a(A, K, mbase, kb, lane);
    v16bf b = load_frag_b_row(W, K, nbase + (lane & 15), kb, lane);
    c = __builtin_amdgcn_wmma_f32_16x16x32_bf16(false, a, false, b, (short)0, c, false, false);
  }
  int n = nbase + (lane & 15);
  int mo = (lane < 16) ? 0 : 8;
  float bv = bias ? bias[n] : 0.0f;
#pragma unroll
  for (int j = 0; j < 8; ++j) {
    int m = mbase + mo + j;
    float v = c[j] + bv;
    if (do_relu) v = fmaxf(v, 0.0f);
    if (resid) v += resid[(size_t)m * Nn + n];
    if (Yf32) Yf32[(size_t)m * Nn + n] = v;
    if (Yb16) Yb16[(size_t)m * Nn + n] = f2bf(v);
  }
}

// Streamed row-wise logsumexp of emission logits resE @ termE^T, never materializing [C][V].
// grid = (C/16, CHUNKS), 1 wave each; A fragments hoisted, online max/sum merge via LDS.
__global__ void emis_lse_partial_kernel(const __bf16* __restrict__ A,
                                        const __bf16* __restrict__ Tm,
                                        float* __restrict__ pmax,
                                        float* __restrict__ psum) {
  int mbase = blockIdx.x << 4;
  int chunk = blockIdx.y;
  int lane = threadIdx.x;
  v16bf afr[8];
#pragma unroll
  for (int kk = 0; kk < 8; ++kk) afr[kk] = load_frag_a(A, HH, mbase, kk * 32, lane);
  __shared__ float tile[16][17];
  float mx = -INFINITY, sm = 0.0f;
  for (int tt = 0; tt < TILES_PER_CHUNK; ++tt) {
    int vbase = (chunk * TILES_PER_CHUNK + tt) << 4;
    v8f c = {};
#pragma unroll
    for (int kk = 0; kk < 8; ++kk) {
      v16bf b = load_frag_b_row(Tm, HH, vbase + (lane & 15), kk * 32, lane);
      c = __builtin_amdgcn_wmma_f32_16x16x32_bf16(false, afr[kk], false, b, (short)0, c, false, false);
    }
    int n = lane & 15;
    int mo = (lane < 16) ? 0 : 8;
#pragma unroll
    for (int j = 0; j < 8; ++j) tile[mo + j][n] = c[j];
    __syncthreads();
    if (lane < 16) {
      float tm = -INFINITY;
#pragma unroll
      for (int q = 0; q < 16; ++q) tm = fmaxf(tm, tile[lane][q]);
      float ts = 0.0f;
#pragma unroll
      for (int q = 0; q < 16; ++q) ts += expf(tile[lane][q] - tm);
      if (tm > mx) { sm = sm * expf(mx - tm) + ts; mx = tm; }
      else         { sm += ts * expf(tm - mx); }
    }
    __syncthreads();
  }
  if (lane < 16) {
    int row = mbase + lane;
    pmax[row * CHUNKS + chunk] = mx;
    psum[row * CHUNKS + chunk] = sm;
  }
}

__global__ void lse_combine_kernel(const float* __restrict__ pmax,
                                   const float* __restrict__ psum,
                                   float* __restrict__ lseE) {
  int c = blockIdx.x * blockDim.x + threadIdx.x;
  if (c >= CC) return;
  float mx = -INFINITY, sm = 0.0f;
  for (int k = 0; k < CHUNKS; ++k) {
    float tm = pmax[c * CHUNKS + k];
    float ts = psum[c * CHUNKS + k];
    if (ts > 0.0f) {
      if (tm > mx) { sm = sm * expf(mx - tm) + ts; mx = tm; }
      else         { sm += ts * expf(tm - mx); }
    }
  }
  lseE[c] = mx + logf(sm);
}

// obsE[p][c] = dot(resE[c,:], termE[text[p],:]) - lseE[c]; WMMA with gathered token rows as B.
__global__ void obs_logits_kernel(const __bf16* __restrict__ A,
                                  const __bf16* __restrict__ Tm,
                                  const int* __restrict__ text,
                                  const float* __restrict__ lseE,
                                  float* __restrict__ obsE) {
  int mbase = blockIdx.x << 4;
  int pbase = blockIdx.y << 4;
  int lane = threadIdx.x;
  int p = pbase + (lane & 15);
  int tok = text[p];
  v8f c = {};
#pragma unroll
  for (int kk = 0; kk < 8; ++kk) {
    v16bf a = load_frag_a(A, HH, mbase, kk * 32, lane);
    v16bf b = load_frag_b_row(Tm, HH, tok, kk * 32, lane);
    c = __builtin_amdgcn_wmma_f32_16x16x32_bf16(false, a, false, b, (short)0, c, false, false);
  }
  int mo = (lane < 16) ? 0 : 8;
#pragma unroll
  for (int j = 0; j < 8; ++j) {
    int ci = mbase + mo + j;
    obsE[(size_t)p * CC + ci] = c[j] - lseE[ci];
  }
}

__global__ void row_logsoftmax_kernel(float* __restrict__ X) {
  int r = blockIdx.x, c = threadIdx.x;
  float v = X[(size_t)r * CC + c];
  __shared__ float red[CC];
  red[c] = v; __syncthreads();
  for (int s = CC / 2; s > 0; s >>= 1) { if (c < s) red[c] = fmaxf(red[c], red[c + s]); __syncthreads(); }
  float mx = red[0]; __syncthreads();
  red[c] = expf(v - mx); __syncthreads();
  for (int s = CC / 2; s > 0; s >>= 1) { if (c < s) red[c] += red[c + s]; __syncthreads(); }
  X[(size_t)r * CC + c] = v - (mx + logf(red[0]));
}

__global__ void start_head_kernel(const float* __restrict__ resS,
                                  const float* __restrict__ soW,
                                  const float* __restrict__ sob,
                                  float* __restrict__ startv) {
  int c = threadIdx.x;
  float acc = 0.0f;
  for (int k = 0; k < HH; ++k) acc += resS[(size_t)c * HH + k] * soW[k];
  acc += sob[0];
  __shared__ float red[CC];
  red[c] = acc; __syncthreads();
  for (int s = CC / 2; s > 0; s >>= 1) { if (c < s) red[c] = fmaxf(red[c], red[c + s]); __syncthreads(); }
  float mx = red[0]; __syncthreads();
  red[c] = expf(acc - mx); __syncthreads();
  for (int s = CC / 2; s > 0; s >>= 1) { if (c < s) red[c] += red[c + s]; __syncthreads(); }
  startv[c] = acc - (mx + logf(red[0]));
}

// alpha scan: one workgroup per batch element, alpha vector kept in LDS.
__global__ void hmm_forward_kernel(const float* __restrict__ startv,
                                   const float* __restrict__ trans,
                                   const float* __restrict__ obsE,
                                   const int* __restrict__ lengths,
                                   float* __restrict__ alphas) {
  int n = blockIdx.x, c = threadIdx.x;
  int len = lengths[n];
  __shared__ float ap[CC];
  float a = startv[c] + obsE[((size_t)n * TT) * CC + c];
  alphas[((size_t)n * TT) * CC + c] = a;
  for (int t = 1; t < TT; ++t) {
    ap[c] = a;
    __syncthreads();
    if (t < len) {
      float mx = -INFINITY;
      for (int i = 0; i < CC; ++i) mx = fmaxf(mx, ap[i] + trans[i * CC + c]);
      float s = 0.0f;
      for (int i = 0; i < CC; ++i) s += expf(ap[i] + trans[i * CC + c] - mx);
      a = mx + logf(s) + obsE[((size_t)n * TT + t) * CC + c];
    }
    alphas[((size_t)n * TT + t) * CC + c] = a;
    __syncthreads();
  }
}

__global__ void hmm_backward_kernel(const float* __restrict__ trans,
                                    const float* __restrict__ obsE,
                                    const int* __restrict__ lengths,
                                    float* __restrict__ betas) {
  int n = blockIdx.x, j = threadIdx.x;
  int len = lengths[n];
  __shared__ float w[CC];
  float b = 0.0f;
  betas[((size_t)n * TT + TT - 1) * CC + j] = 0.0f;
  for (int t = TT - 2; t >= 0; --t) {
    w[j] = obsE[((size_t)n * TT + t + 1) * CC + j] + b;
    __syncthreads();
    if (t + 1 < len) {
      float mx = -INFINITY;
      for (int i = 0; i < CC; ++i) mx = fmaxf(mx, trans[j * CC + i] + w[i]);
      float s = 0.0f;
      for (int i = 0; i < CC; ++i) s += expf(trans[j * CC + i] + w[i] - mx);
      b = mx + logf(s);
    }
    betas[((size_t)n * TT + t) * CC + j] = b;
    __syncthreads();
  }
}

__global__ void zero_kernel(float* __restrict__ p, int n) {
  int i = blockIdx.x * blockDim.x + threadIdx.x;
  if (i < n) p[i] = 0.0f;
}

__global__ void evidence_kernel(const float* __restrict__ alphas,
                                const int* __restrict__ lengths,
                                float* __restrict__ Z, float* __restrict__ accum) {
  int n = blockIdx.x, c = threadIdx.x;
  int t = lengths[n] - 1;
  float v = alphas[((size_t)n * TT + t) * CC + c];
  __shared__ float red[CC];
  red[c] = v; __syncthreads();
  for (int s = CC / 2; s > 0; s >>= 1) { if (c < s) red[c] = fmaxf(red[c], red[c + s]); __syncthreads(); }
  float mx = red[0]; __syncthreads();
  red[c] = expf(v - mx); __syncthreads();
  for (int s = CC / 2; s > 0; s >>= 1) { if (c < s) red[c] += red[c + s]; __syncthreads(); }
  if (c == 0) {
    float z = mx + logf(red[0]);
    Z[n] = z;
    atomicAdd(&accum[1], z);
  }
}

// elbo = sum over (n,t,i,j) of m * P; one block per (n,t), thread j loops i.
__global__ void elbo_kernel(const float* __restrict__ alphas,
                            const float* __restrict__ betas,
                            const float* __restrict__ obsE,
                            const float* __restrict__ trans,
                            const float* __restrict__ startv,
                            const float* __restrict__ Z,
                            const int* __restrict__ lengths,
                            float* __restrict__ accum) {
  int blk = blockIdx.x;
  int n = blk / (TT - 1);
  int t = blk % (TT - 1);
  if (t + 1 >= lengths[n]) return;  // masked step contributes 0 (uniform exit)
  int j = threadIdx.x;
  __shared__ float w[CC];   // obsE[n,t+1,i] + beta[n,t+1,i]
  __shared__ float w2[CC];  // obsE[n,t+1,i]
  __shared__ float red[CC];
  float o = obsE[((size_t)n * TT + t + 1) * CC + j];
  w2[j] = o;
  w[j] = o + betas[((size_t)n * TT + t + 1) * CC + j];
  __syncthreads();
  float aj = alphas[((size_t)n * TT + t) * CC + j];
  float zb = Z[n];
  float s0 = (t == 0) ? (startv[j] + obsE[((size_t)n * TT) * CC + j]) : 0.0f;
  float acc = 0.0f;
  for (int i = 0; i < CC; ++i) {
    float tr = trans[j * CC + i];
    acc += expf(aj + tr + w[i] - zb) * (tr + w2[i] + s0);
  }
  red[j] = acc; __syncthreads();
  for (int s = CC / 2; s > 0; s >>= 1) { if (j < s) red[j] += red[j + s]; __syncthreads(); }
  if (j == 0) atomicAdd(&accum[0], red[0]);
}

__global__ void write_out_kernel(const float* __restrict__ accum, float* __restrict__ out) {
  if (threadIdx.x == 0) { out[0] = accum[0]; out[1] = accum[1]; }
}

extern "C" void kernel_launch(void* const* d_in, const int* in_sizes, int n_in,
                              void* d_out, int out_size, void* d_ws, size_t ws_size,
                              hipStream_t stream) {
  (void)in_sizes; (void)n_in; (void)out_size; (void)ws_size;
  // setup_inputs() dict order
  const int*   text     = (const int*)d_in[0];
  const int*   lengths  = (const int*)d_in[1];
  // d_in[2] = mask (recomputed from lengths on device)
  const float* start_emb = (const float*)d_in[3];
  const float* state_emb = (const float*)d_in[4];
  const float* next_emb  = (const float*)d_in[5];
  const float* pret_emb  = (const float*)d_in[6];
  const float* term_emb  = (const float*)d_in[7];
  const float* sW1 = (const float*)d_in[8];  const float* sb1 = (const float*)d_in[9];
  const float* sW2 = (const float*)d_in[10]; const float* sb2 = (const float*)d_in[11];
  const float* tW1 = (const float*)d_in[12]; const float* tb1 = (const float*)d_in[13];
  const float* tW2 = (const float*)d_in[14]; const float* tb2 = (const float*)d_in[15];
  const float* eW1 = (const float*)d_in[16]; const float* eb1 = (const float*)d_in[17];
  const float* eW2 = (const float*)d_in[18]; const float* eb2 = (const float*)d_in[19];
  const float* soW = (const float*)d_in[20]; const float* sob = (const float*)d_in[21];

  char* base = (char*)d_ws;
  size_t off = 0;
  auto alloc = [&](size_t bytes) -> void* {
    void* p = base + off;
    off = (off + bytes + 255) & ~(size_t)255;
    return p;
  };

  const size_t MB = (size_t)CC * HH;  // 65536
  __bf16* sW1b   = (__bf16*)alloc(MB * 2);
  __bf16* sW2b   = (__bf16*)alloc(MB * 2);
  __bf16* tW1b   = (__bf16*)alloc(MB * 2);
  __bf16* tW2b   = (__bf16*)alloc(MB * 2);
  __bf16* eW1b   = (__bf16*)alloc(MB * 2);
  __bf16* eW2b   = (__bf16*)alloc(MB * 2);
  __bf16* startEb= (__bf16*)alloc(MB * 2);
  __bf16* stateEb= (__bf16*)alloc(MB * 2);
  __bf16* pretEb = (__bf16*)alloc(MB * 2);
  __bf16* nextEb = (__bf16*)alloc(MB * 2);
  __bf16* termEb = (__bf16*)alloc((size_t)VV * HH * 2);
  __bf16* hB16   = (__bf16*)alloc(MB * 2);
  __bf16* resTb  = (__bf16*)alloc(MB * 2);
  __bf16* resEb  = (__bf16*)alloc(MB * 2);

  float* hF32   = (float*)alloc(MB * 4);
  float* resSf  = (float*)alloc(MB * 4);
  float* trans  = (float*)alloc((size_t)CC * CC * 4);
  float* startv = (float*)alloc(CC * 4);
  float* lseE   = (float*)alloc(CC * 4);
  float* pmax   = (float*)alloc((size_t)CC * CHUNKS * 4);
  float* psum   = (float*)alloc((size_t)CC * CHUNKS * 4);
  float* obsE   = (float*)alloc((size_t)PP * CC * 4);
  float* alphas = (float*)alloc((size_t)PP * CC * 4);
  float* betas  = (float*)alloc((size_t)PP * CC * 4);
  float* Zbuf   = (float*)alloc(NN * 4);
  float* accum  = (float*)alloc(2 * 4);

  auto cvt = [&](const float* s, __bf16* d, int n) {
    cvt_f32_to_bf16_kernel<<<(n + 255) / 256, 256, 0, stream>>>(s, d, n);
  };
  cvt(sW1, sW1b, MB);  cvt(sW2, sW2b, MB);
  cvt(tW1, tW1b, MB);  cvt(tW2, tW2b, MB);
  cvt(eW1, eW1b, MB);  cvt(eW2, eW2b, MB);
  cvt(start_emb, startEb, MB);
  cvt(state_emb, stateEb, MB);
  cvt(pret_emb,  pretEb,  MB);
  cvt(next_emb,  nextEb,  MB);
  cvt(term_emb,  termEb,  VV * HH);

  zero_kernel<<<1, 32, 0, stream>>>(accum, 2);

  // Residual MLPs: h = relu(X@W1^T + b1); y = relu(h@W2^T + b2) + h
  gemm256_bf16_wmma_kernel<<<256, 32, 0, stream>>>(startEb, sW1b, sb1, nullptr, hF32, hB16, HH, HH, 1);
  gemm256_bf16_wmma_kernel<<<256, 32, 0, stream>>>(hB16,    sW2b, sb2, hF32, resSf, nullptr, HH, HH, 1);
  gemm256_bf16_wmma_kernel<<<256, 32, 0, stream>>>(stateEb, tW1b, tb1, nullptr, hF32, hB16, HH, HH, 1);
  gemm256_bf16_wmma_kernel<<<256, 32, 0, stream>>>(hB16,    tW2b, tb2, hF32, nullptr, resTb, HH, HH, 1);
  gemm256_bf16_wmma_kernel<<<256, 32, 0, stream>>>(pretEb,  eW1b, eb1, nullptr, hF32, hB16, HH, HH, 1);
  gemm256_bf16_wmma_kernel<<<256, 32, 0, stream>>>(hB16,    eW2b, eb2, hF32, nullptr, resEb, HH, HH, 1);

  // transition logits + row log-softmax
  gemm256_bf16_wmma_kernel<<<256, 32, 0, stream>>>(resTb, nextEb, nullptr, nullptr, trans, nullptr, CC, HH, 0);
  row_logsoftmax_kernel<<<CC, CC, 0, stream>>>(trans);

  // start distribution
  start_head_kernel<<<1, CC, 0, stream>>>(resSf, soW, sob, startv);

  // emission logsumexp (streamed over V) + gathered obsE columns
  emis_lse_partial_kernel<<<dim3(CC / 16, CHUNKS), 32, 0, stream>>>(resEb, termEb, pmax, psum);
  lse_combine_kernel<<<1, CC, 0, stream>>>(pmax, psum, lseE);
  obs_logits_kernel<<<dim3(CC / 16, PP / 16), 32, 0, stream>>>(resEb, termEb, text, lseE, obsE);

  // forward-backward, evidence, elbo
  hmm_forward_kernel<<<NN, CC, 0, stream>>>(startv, trans, obsE, lengths, alphas);
  hmm_backward_kernel<<<NN, CC, 0, stream>>>(trans, obsE, lengths, betas);
  evidence_kernel<<<NN, CC, 0, stream>>>(alphas, lengths, Zbuf, accum);
  elbo_kernel<<<NN * (TT - 1), CC, 0, stream>>>(alphas, betas, obsE, trans, startv, Zbuf, lengths, accum);

  write_out_kernel<<<1, 32, 0, stream>>>(accum, (float*)d_out);
}
